// Reconstruct_model_21045339750972
// MI455X (gfx1250) — matrix operand order repaired and checked by reading, not compiled, not run on hardware
//
#include <hip/hip_runtime.h>
#include <math.h>

#define IN_DIMC 128
#define HIDF    64
#define QKD     32

typedef __attribute__((ext_vector_type(16))) _Float16 v16h;
typedef __attribute__((ext_vector_type(8)))  _Float16 v8h;
typedef __attribute__((ext_vector_type(8)))  float    v8f;
typedef __attribute__((ext_vector_type(4)))  float    v4f;

// ---------------------------------------------------------------------------
// WMMA helpers (CDNA5 wave32, 16x16x32 f16 -> f32)
// ---------------------------------------------------------------------------
static __device__ __forceinline__ v8f wmma16(v16h a, v16h b, v8f c) {
  // 8 args: (neg_a, A, neg_b, B, c_mod, C, reuse_a, reuse_b)
  return __builtin_amdgcn_wmma_f32_16x16x32_f16(false, a, false, b, (short)0, c,
                                                false, false);
}

// A-matrix (16x32 f16) per-lane layout: lane half kh holds K in
// [16*kh, 16*kh+16) (elements 0..15).
// `base` points at row start + 32*ktile in a row-major f32 matrix.
static __device__ __forceinline__ v16h load_a_f32(const float* base, int kh) {
  const v4f* p0 = (const v4f*)(base + 16 * kh);
  const v4f* p1 = (const v4f*)(base + 16 * kh + 8);
  v4f l0 = p0[0], l1 = p0[1];
  v4f h0 = p1[0], h1 = p1[1];
  v16h a;
#pragma unroll
  for (int e = 0; e < 4; ++e) {
    a[e]      = (_Float16)l0[e];
    a[4 + e]  = (_Float16)l1[e];
    a[8 + e]  = (_Float16)h0[e];
    a[12 + e] = (_Float16)h1[e];
  }
  return a;
}

// Same, but source already f16 row-major (global or LDS).
static __device__ __forceinline__ v16h load_a_f16(const _Float16* base, int kh) {
  v8h lo = *(const v8h*)(base + 16 * kh);
  v8h hi = *(const v8h*)(base + 16 * kh + 8);
  v16h a;
#pragma unroll
  for (int e = 0; e < 8; ++e) {
    a[e]     = lo[e];
    a[8 + e] = hi[e];
  }
  return a;
}

// ---------------------------------------------------------------------------
// Pack a row-major f32 KxNn weight into WMMA-B register order:
//   dst[(((kt*Nt+nt)*32 + lane)*16) + e] = W[(32*kt + 16*(lane>>4) + e)*Nn
//                                           + 16*nt + (lane&15)]
// so the B operand is one contiguous 32B v16h load per lane.
// ---------------------------------------------------------------------------
__global__ void pack_b_f16(const float* __restrict__ W, _Float16* __restrict__ dst,
                           int K, int Nn) {
  int o = blockIdx.x * blockDim.x + threadIdx.x;
  if (o >= K * Nn) return;
  int e    = o & 15;
  int lane = (o >> 4) & 31;
  int tile = o >> 9;
  int Nt   = Nn >> 4;
  int kt   = tile / Nt;
  int nt   = tile - kt * Nt;
  int k    = 32 * kt + 16 * (lane >> 4) + e;
  int nn   = 16 * nt + (lane & 15);
  dst[o] = (_Float16)W[k * Nn + nn];
}

// ---------------------------------------------------------------------------
// Encoder: h = relu(relu(X W1 + b1) W2 + b2) ; one wave = 16 rows.
// blockIdx.y selects (x_train->h_train) or (x_ori->h_ori).
// ---------------------------------------------------------------------------
#define ENC_WAVES 8
__global__ __launch_bounds__(256) void encoder_kernel(
    const float* __restrict__ x_train, const float* __restrict__ x_ori,
    const _Float16* __restrict__ W1p, const float* __restrict__ b1,
    const _Float16* __restrict__ W2p, const float* __restrict__ b2,
    float* __restrict__ h_train, float* __restrict__ h_ori) {
  __shared__ __align__(16) _Float16 lds[ENC_WAVES][16][HIDF];
  const int lane = threadIdx.x & 31;
  const int wv   = threadIdx.x >> 5;
  const int rowTile = blockIdx.x * ENC_WAVES + wv;
  const float* X = blockIdx.y ? x_ori : x_train;
  float*       H = blockIdx.y ? h_ori : h_train;
  const int m  = lane & 15;
  const int kh = lane >> 4;
  const size_t row = (size_t)rowTile * 16 + m;

  // Layer 1: (16x128)*(128x64)
  v8f c[4] = {};
#pragma unroll
  for (int kt = 0; kt < 4; ++kt) {
    v16h a = load_a_f32(X + row * IN_DIMC + 32 * kt, kh);
#pragma unroll
    for (int nt = 0; nt < 4; ++nt) {
      v16h b = *(const v16h*)(W1p + (((kt * 4 + nt) * 32 + lane) << 4));
      c[nt] = wmma16(a, b, c[nt]);
    }
  }
  // relu + bias, reshape C-layout -> row-major f16 in LDS
#pragma unroll
  for (int nt = 0; nt < 4; ++nt) {
#pragma unroll
    for (int r = 0; r < 8; ++r) {
      int rr   = (lane < 16) ? r : r + 8;
      int colg = 16 * nt + m;
      float v  = c[nt][r] + b1[colg];
      lds[wv][rr][colg] = (_Float16)(v > 0.f ? v : 0.f);
    }
  }
  __syncthreads();

  // Layer 2: (16x64)*(64x64)
  v8f d[4] = {};
#pragma unroll
  for (int kt = 0; kt < 2; ++kt) {
    v16h a = load_a_f16(&lds[wv][m][32 * kt], kh);
#pragma unroll
    for (int nt = 0; nt < 4; ++nt) {
      v16h b = *(const v16h*)(W2p + (((kt * 4 + nt) * 32 + lane) << 4));
      d[nt] = wmma16(a, b, d[nt]);
    }
  }
#pragma unroll
  for (int nt = 0; nt < 4; ++nt) {
#pragma unroll
    for (int r = 0; r < 8; ++r) {
      int rr   = (lane < 16) ? r : r + 8;
      int colg = 16 * nt + m;
      float v  = d[nt][r] + b2[colg];
      H[((size_t)rowTile * 16 + rr) * HIDF + colg] = v > 0.f ? v : 0.f;
    }
  }
}

// ---------------------------------------------------------------------------
// Graph helpers
// ---------------------------------------------------------------------------
__global__ void zero_f32(float* __restrict__ p, size_t n) {
  size_t i = (size_t)blockIdx.x * blockDim.x + threadIdx.x;
  if (i < n) p[i] = 0.f;
}

__global__ void degree_kernel(const long long* __restrict__ row,
                              float* __restrict__ deg, int E) {
  int e = blockIdx.x * blockDim.x + threadIdx.x;
  if (e < E) atomicAdd(&deg[row[e]], 1.0f);
}

__global__ void dinv_kernel(float* __restrict__ deg, int n) {
  int i = blockIdx.x * blockDim.x + threadIdx.x;
  if (i < n) {
    float d = deg[i];
    d = d < 1.f ? 1.f : d;
    deg[i] = 1.0f / sqrtf(d);   // becomes d_invsqrt in place
  }
}

// tmp[col] += x[row]*dinv[row], 64 features per edge
__global__ void scatter_kernel(const long long* __restrict__ row,
                               const long long* __restrict__ col,
                               const float* __restrict__ x,
                               const float* __restrict__ dinv,
                               float* __restrict__ tmp, int E) {
  size_t idx = (size_t)blockIdx.x * blockDim.x + threadIdx.x;
  int e = (int)(idx >> 6);
  int f = (int)(idx & 63);
  if (e < E) {
    long long r = row[e], c = col[e];
    atomicAdd(&tmp[(size_t)c * HIDF + f], x[(size_t)r * HIDF + f] * dinv[r]);
  }
}

// lap = x - tmp * dinv[node]
__global__ void lap_combine(const float* __restrict__ x,
                            const float* __restrict__ tmp,
                            const float* __restrict__ dinv,
                            float* __restrict__ lap, size_t n64) {
  size_t i = (size_t)blockIdx.x * blockDim.x + threadIdx.x;
  if (i < n64) lap[i] = x[i] - tmp[i] * dinv[i >> 6];
}

// masked column sum of h_ori (64 cols) + weight count
__global__ void colsum_kernel(const float* __restrict__ h_ori,
                              const unsigned char* __restrict__ mask,
                              float* __restrict__ acc, float* __restrict__ wsum,
                              int n) {
  int tid   = blockIdx.x * blockDim.x + threadIdx.x;
  int f     = tid & 63;
  int chunk = tid >> 6;
  int r0 = chunk * 64;
  if (r0 >= n) return;
  int r1 = r0 + 64; if (r1 > n) r1 = n;
  float s = 0.f, cnt = 0.f;
  for (int r = r0; r < r1; ++r) {
    float w = mask[r] ? 0.f : 1.f;
    s += w * h_ori[(size_t)r * HIDF + f];
    cnt += w;
  }
  atomicAdd(&acc[f], s);
  if (f == 0) atomicAdd(wsum, cnt);
}

// q_global = (acc @ Wq + bq*wsum)/max(wsum,1); qk = Wk q_global; sb = q·bk
__global__ void qk_kernel(const float* __restrict__ acc,
                          const float* __restrict__ wsum_p,
                          const float* __restrict__ Wq, const float* __restrict__ bq,
                          const float* __restrict__ Wk, const float* __restrict__ bk,
                          float* __restrict__ qkvec, float* __restrict__ sb) {
  __shared__ float qg[QKD];
  int t = threadIdx.x;
  float wsum = *wsum_p;
  float W = wsum < 1.f ? 1.f : wsum;
  if (t < QKD) {
    float s = 0.f;
    for (int f = 0; f < HIDF; ++f) s += acc[f] * Wq[f * QKD + t];
    qg[t] = (s + bq[t] * wsum) / W;
  }
  __syncthreads();
  if (t < HIDF) {
    float s = 0.f;
    for (int q = 0; q < QKD; ++q) s += Wk[t * QKD + q] * qg[q];
    qkvec[t] = s;
  }
  if (t == 0) {
    float s = 0.f;
    for (int q = 0; q < QKD; ++q) s += qg[q] * bk[q];
    *sb = s;
  }
}

// ---------------------------------------------------------------------------
// Per-node attention: scores_k = (qk·L_k + sb)/8 ; softmax(3) ; Lc = ΣA_k L_k
// One wave per node, lane handles features l and l+32.
// ---------------------------------------------------------------------------
__global__ __launch_bounds__(32) void attn_kernel(
    const float* __restrict__ h, const float* __restrict__ lap1,
    const float* __restrict__ lap2, const float* __restrict__ qkvec,
    const float* __restrict__ sbp, _Float16* __restrict__ Lc) {
  const int node = blockIdx.x;
  const int l = threadIdx.x;
  const size_t base = (size_t)node * HIDF;
  float ha = h[base + l],    hb = h[base + l + 32];
  float l1a = lap1[base + l], l1b = lap1[base + l + 32];
  float l2a = lap2[base + l], l2b = lap2[base + l + 32];
  float L0a = 2.f * ha - l1a, L0b = 2.f * hb - l1b;
  float qa = qkvec[l], qb = qkvec[l + 32];

  float s0 = qa * L0a + qb * L0b;
  float s1 = qa * l1a + qb * l1b;
  float s2 = qa * l2a + qb * l2b;
#pragma unroll
  for (int o = 16; o > 0; o >>= 1) {
    s0 += __shfl_xor(s0, o, 32);
    s1 += __shfl_xor(s1, o, 32);
    s2 += __shfl_xor(s2, o, 32);
  }
  const float sb = *sbp;
  const float scale = 0.125f;  // 1/sqrt(64)
  s0 = (s0 + sb) * scale;
  s1 = (s1 + sb) * scale;
  s2 = (s2 + sb) * scale;
  float mx = fmaxf(s0, fmaxf(s1, s2));
  float e0 = expf(s0 - mx), e1 = expf(s1 - mx), e2 = expf(s2 - mx);
  float inv = 1.f / (e0 + e1 + e2);
  float A0 = e0 * inv, A1 = e1 * inv, A2 = e2 * inv;
  Lc[base + l]      = (_Float16)(A0 * L0a + A1 * l1a + A2 * l2a);
  Lc[base + l + 32] = (_Float16)(A0 * L0b + A1 * l1b + A2 * l2b);
}

// ---------------------------------------------------------------------------
// z = Lc @ Wv + bv  (N x 64), stored as f16 for the big GEMMs.
// ---------------------------------------------------------------------------
__global__ __launch_bounds__(256) void zmat_kernel(
    const _Float16* __restrict__ Lc, const _Float16* __restrict__ Wvp,
    const float* __restrict__ bv, _Float16* __restrict__ zh) {
  const int lane = threadIdx.x & 31, wv = threadIdx.x >> 5;
  const int rowTile = blockIdx.x * 8 + wv;
  const int m = lane & 15, kh = lane >> 4;
  v8f c[4] = {};
#pragma unroll
  for (int kt = 0; kt < 2; ++kt) {
    v16h a = load_a_f16(Lc + ((size_t)rowTile * 16 + m) * HIDF + 32 * kt, kh);
#pragma unroll
    for (int nt = 0; nt < 4; ++nt) {
      v16h b = *(const v16h*)(Wvp + (((kt * 4 + nt) * 32 + lane) << 4));
      c[nt] = wmma16(a, b, c[nt]);
    }
  }
#pragma unroll
  for (int nt = 0; nt < 4; ++nt) {
#pragma unroll
    for (int r = 0; r < 8; ++r) {
      int rr = (lane < 16) ? r : r + 8;
      int colg = 16 * nt + m;
      zh[((size_t)rowTile * 16 + rr) * HIDF + colg] = (_Float16)(c[nt][r] + bv[colg]);
    }
  }
}

// ---------------------------------------------------------------------------
// x_hat = z @ Wdx + bdx  (N x 128). Output written once -> non-temporal.
// ---------------------------------------------------------------------------
__global__ __launch_bounds__(256) void xhat_kernel(
    const _Float16* __restrict__ zh, const _Float16* __restrict__ Wdxp,
    const float* __restrict__ bdx, float* __restrict__ out) {
  const int lane = threadIdx.x & 31, wv = threadIdx.x >> 5;
  const int rowTile = blockIdx.x * 8 + wv;
  const int m = lane & 15, kh = lane >> 4;
  v8f c[8] = {};
#pragma unroll
  for (int kt = 0; kt < 2; ++kt) {
    v16h a = load_a_f16(zh + ((size_t)rowTile * 16 + m) * HIDF + 32 * kt, kh);
#pragma unroll
    for (int nt = 0; nt < 8; ++nt) {
      v16h b = *(const v16h*)(Wdxp + (((kt * 8 + nt) * 32 + lane) << 4));
      c[nt] = wmma16(a, b, c[nt]);
    }
  }
#pragma unroll
  for (int nt = 0; nt < 8; ++nt) {
#pragma unroll
    for (int r = 0; r < 8; ++r) {
      int rr = (lane < 16) ? r : r + 8;
      int colg = 16 * nt + m;
      __builtin_nontemporal_store(
          c[nt][r] + bdx[colg],
          out + ((size_t)rowTile * 16 + rr) * IN_DIMC + colg);
    }
  }
}

// ---------------------------------------------------------------------------
// adj = z z^T  (N x N). One wave computes a 16x64 strip (8 WMMAs).
// B columns of z^T are rows of z -> contiguous v16h loads (all of z fits in
// L2, 1.5 MB). The 604 MB output streams out with non-temporal stores so it
// never evicts reused data from L2.
// ---------------------------------------------------------------------------
__global__ __launch_bounds__(256) void adj_kernel(
    const _Float16* __restrict__ zh, float* __restrict__ adj, int n) {
  const int lane = threadIdx.x & 31, wv = threadIdx.x >> 5;
  const int rowTile = blockIdx.y * 8 + wv;       // n/16 tiles
  const int colBase = blockIdx.x * 64;
  const int m = lane & 15, kh = lane >> 4;
  const size_t arow = (size_t)rowTile * 16 + m;
  v16h a0 = load_a_f16(zh + arow * HIDF, kh);
  v16h a1 = load_a_f16(zh + arow * HIDF + 32, kh);
#pragma unroll
  for (int nt = 0; nt < 4; ++nt) {
    const size_t zrow = (size_t)colBase + 16 * nt + m;   // B column = lane&15
    if (colBase + 64 < n)
      __builtin_prefetch(zh + (zrow + 64) * HIDF, 0, 0); // global_prefetch hint
    v16h b0 = *(const v16h*)(zh + zrow * HIDF + 16 * kh);
    v16h b1 = *(const v16h*)(zh + zrow * HIDF + 32 + 16 * kh);
    v8f c = {};
    c = wmma16(a0, b0, c);
    c = wmma16(a1, b1, c);
#pragma unroll
    for (int r = 0; r < 8; ++r) {
      int rr = (lane < 16) ? r : r + 8;
      __builtin_nontemporal_store(
          c[r], adj + ((size_t)rowTile * 16 + rr) * (size_t)n + zrow);
    }
  }
}

// ---------------------------------------------------------------------------
// Host orchestration
// ---------------------------------------------------------------------------
extern "C" void kernel_launch(void* const* d_in, const int* in_sizes, int n_in,
                              void* d_out, int out_size, void* d_ws, size_t ws_size,
                              hipStream_t stream) {
  const float*     x_train = (const float*)d_in[0];
  const float*     x_ori   = (const float*)d_in[1];
  const long long* eidx    = (const long long*)d_in[2];
  const unsigned char* mask = (const unsigned char*)d_in[3];
  const float* W1  = (const float*)d_in[4];
  const float* b1  = (const float*)d_in[5];
  const float* W2  = (const float*)d_in[6];
  const float* b2  = (const float*)d_in[7];
  const float* Wq  = (const float*)d_in[8];
  const float* bq  = (const float*)d_in[9];
  const float* Wk  = (const float*)d_in[10];
  const float* bk  = (const float*)d_in[11];
  const float* Wv  = (const float*)d_in[12];
  const float* bv  = (const float*)d_in[13];
  const float* Wdx = (const float*)d_in[14];
  const float* bdx = (const float*)d_in[15];

  const int n = in_sizes[0] / IN_DIMC;   // 12288
  const int E = in_sizes[2] / 2;         // 393216
  const long long* row = eidx;
  const long long* col = eidx + E;

  float* out_xhat = (float*)d_out;
  float* out_adj  = out_xhat + (size_t)n * IN_DIMC;

  // workspace carve-out (256B aligned segments)
  char* ws = (char*)d_ws;
  size_t off = 0;
  auto take = [&](size_t bytes) -> void* {
    void* p = ws + off;
    off = (off + bytes + 255) & ~(size_t)255;
    return p;
  };
  float*    h_train = (float*)take((size_t)n * HIDF * 4);
  float*    h_ori   = (float*)take((size_t)n * HIDF * 4);
  float*    dinv    = (float*)take((size_t)n * 4);       // deg then d^-1/2
  float*    tmp     = (float*)take((size_t)n * HIDF * 4);
  float*    lap1    = (float*)take((size_t)n * HIDF * 4);
  float*    lap2    = (float*)take((size_t)n * HIDF * 4);
  _Float16* Lc      = (_Float16*)take((size_t)n * HIDF * 2);
  _Float16* zh      = (_Float16*)take((size_t)n * HIDF * 2);
  float*    acc     = (float*)take(65 * 4);              // 64 colsums + wsum
  float*    wsum    = acc + 64;
  float*    qkvec   = (float*)take(HIDF * 4);
  float*    sb      = (float*)take(4);
  _Float16* W1p     = (_Float16*)take(IN_DIMC * HIDF * 2);
  _Float16* W2p     = (_Float16*)take(HIDF * HIDF * 2);
  _Float16* Wvp     = (_Float16*)take(HIDF * HIDF * 2);
  _Float16* Wdxp    = (_Float16*)take(HIDF * IN_DIMC * 2);

  const size_t n64 = (size_t)n * HIDF;
  const int rowTiles8 = n / 128;         // waves-of-16-rows / 8 per block

  // 1) pack weights into WMMA-B register order
  pack_b_f16<<<(IN_DIMC * HIDF + 255) / 256, 256, 0, stream>>>(W1, W1p, IN_DIMC, HIDF);
  pack_b_f16<<<(HIDF * HIDF + 255) / 256, 256, 0, stream>>>(W2, W2p, HIDF, HIDF);
  pack_b_f16<<<(HIDF * HIDF + 255) / 256, 256, 0, stream>>>(Wv, Wvp, HIDF, HIDF);
  pack_b_f16<<<(HIDF * IN_DIMC + 255) / 256, 256, 0, stream>>>(Wdx, Wdxp, HIDF, IN_DIMC);

  // 2) encoder for both inputs
  encoder_kernel<<<dim3(rowTiles8, 2), 256, 0, stream>>>(
      x_train, x_ori, W1p, b1, W2p, b2, h_train, h_ori);

  // 3) degrees -> d^{-1/2}
  zero_f32<<<(n + 255) / 256, 256, 0, stream>>>(dinv, (size_t)n);
  degree_kernel<<<(E + 255) / 256, 256, 0, stream>>>(row, dinv, E);
  dinv_kernel<<<(n + 255) / 256, 256, 0, stream>>>(dinv, n);

  // 4) lap1 = lap(h_train)
  zero_f32<<<(unsigned)((n64 + 255) / 256), 256, 0, stream>>>(tmp, n64);
  scatter_kernel<<<(unsigned)(((size_t)E * 64 + 255) / 256), 256, 0, stream>>>(
      row, col, h_train, dinv, tmp, E);
  lap_combine<<<(unsigned)((n64 + 255) / 256), 256, 0, stream>>>(h_train, tmp, dinv, lap1, n64);

  // 5) lap2 = lap(lap1)
  zero_f32<<<(unsigned)((n64 + 255) / 256), 256, 0, stream>>>(tmp, n64);
  scatter_kernel<<<(unsigned)(((size_t)E * 64 + 255) / 256), 256, 0, stream>>>(
      row, col, lap1, dinv, tmp, E);
  lap_combine<<<(unsigned)((n64 + 255) / 256), 256, 0, stream>>>(lap1, tmp, dinv, lap2, n64);

  // 6) q_global -> qk, sb
  zero_f32<<<1, 256, 0, stream>>>(acc, 65);
  {
    int chunks = (n + 63) / 64;
    colsum_kernel<<<(chunks * 64 + 255) / 256, 256, 0, stream>>>(h_ori, mask, acc, wsum, n);
  }
  qk_kernel<<<1, 64, 0, stream>>>(acc, wsum, Wq, bq, Wk, bk, qkvec, sb);

  // 7) attention -> Lc (f16)
  attn_kernel<<<n, 32, 0, stream>>>(h_train, lap1, lap2, qkvec, sb, Lc);

  // 8) z = Lc @ Wv + bv (f16)
  zmat_kernel<<<rowTiles8, 256, 0, stream>>>(Lc, Wvp, bv, zh);

  // 9) x_hat = z @ Wdx + bdx
  xhat_kernel<<<rowTiles8, 256, 0, stream>>>(zh, Wdxp, bdx, out_xhat);

  // 10) adj_hat = z z^T  (the big one: 604 MB of NT stores)
  adj_kernel<<<dim3(n / 64, n / 128), 256, 0, stream>>>(zh, out_adj, n);
}
